// DeepLSTMDecoderLayer_5016521802069
// MI455X (gfx1250) — matrix operand order repaired and checked by reading, not compiled, not run on hardware
//
#include <hip/hip_runtime.h>

// ---------------------------------------------------------------------------
// DeepLSTMDecoderLayer for MI455X (gfx1250): bf16 WMMA everywhere, fp32 accum.
// T=256, S=128, B=32, H=1024, NH=16, D=64, F=4096.
// Round 2: stage straight-copy LDS tiles with GLOBAL_LOAD_ASYNC_TO_LDS_B128
// (ASYNCcnt path) instead of global->VGPR->ds_store.
// ---------------------------------------------------------------------------

#define TT 256
#define SS 128
#define BB 32
#define HH 1024
#define NHH 16
#define DD 64
#define FF 4096

typedef __attribute__((ext_vector_type(16))) __bf16 bf16x16;
typedef __attribute__((ext_vector_type(8)))  float  f32x8;

__device__ __forceinline__ bf16x16 make_frag(uint4 a, uint4 b) {
  union { uint4 u[2]; bf16x16 v; } x;
  x.u[0] = a; x.u[1] = b;
  return x.v;
}

__device__ __forceinline__ f32x8 zero8() {
  f32x8 z = {0.f, 0.f, 0.f, 0.f, 0.f, 0.f, 0.f, 0.f};
  return z;
}

// Async copy of 16B global -> LDS. Per CDNA5 ISA the
// instruction adds INST_OFFSET to BOTH the LDS and the global address, so one
// (lds, gaddr) base pair serves several contiguous chunks via offset:N.
// LDS byte offset = low 32 bits of the generic pointer (ISA §10.2 aperture).
#define ASYNC_B128(lds_u32, gaddr_u64, OFF_LIT)                                \
  asm volatile("global_load_async_to_lds_b128 %0, %1, off offset:" #OFF_LIT    \
               :: "v"(lds_u32), "v"(gaddr_u64) : "memory")

__device__ __forceinline__ void wait_asynccnt0() {
  asm volatile("s_wait_asynccnt 0x0" ::: "memory");
}

// ---------------------------------------------------------------------------
// fp32 -> bf16 conversion
// ---------------------------------------------------------------------------
__global__ void cvt_f32_bf16_kernel(const float* __restrict__ in,
                                    __bf16* __restrict__ out, int n) {
  int i = blockIdx.x * 256 + threadIdx.x;
  if (i < n) out[i] = (__bf16)in[i];
}

// zero c0 and the h-slot of xs[t=0]
__global__ void init_state_kernel(float* __restrict__ c, __bf16* __restrict__ xs) {
  int idx = blockIdx.x * 256 + threadIdx.x;   // 0 .. 32767
  int rb  = idx >> 10;
  int col = idx & 1023;
  c[idx] = 0.0f;
  xs[(size_t)rb * 3072 + 2048 + col] = (__bf16)0.0f;
}

// ---------------------------------------------------------------------------
// Generic bf16 WMMA GEMM:  C[M,N] = act( (A[M,K] @ B[K,N]) + bias ) * alpha
// A row-major (lda), B row-major (ldb). BM=128, BN=128, BK=32.
// 256 threads = 8 waves in 4x2 grid; each wave: 32x64 patch (2x4 WMMA tiles).
// Fragment layouts follow CDNA5 ISA §7.12.2:
//   A (16x32 bf16): lane<16 holds K{0..7,16..23}, lane>=16 K{8..15,24..31}
//   B (32x16 bf16): lane holds column n=lane&15, same K split as A
//   C (16x16 f32) : vgpr r -> row r (lanes 0-15) / r+8 (lanes 16-31), n=lane&15
// ---------------------------------------------------------------------------
template <int ACT, bool OUT_BF16>
__global__ __launch_bounds__(256)
void gemm_bf16_wmma_kernel(const __bf16* __restrict__ A, int lda,
                           const __bf16* __restrict__ B, int ldb,
                           const float* __restrict__ bias, float alpha,
                           void* __restrict__ Cout, int ldc,
                           int M, int N, int K) {
  __shared__ __bf16 As[128 * 40];   // [m][k], row stride 40 (16B aligned, conflict-free)
  __shared__ __bf16 Bt[128 * 40];   // [n][k] (transposed tile)

  const int tid  = threadIdx.x;
  const int lane = tid & 31;
  const int wave = tid >> 5;
  const int wm   = wave >> 1;               // 0..3
  const int wn   = wave & 1;                // 0..1
  const int bn0  = blockIdx.x * 128;
  const int bm0  = blockIdx.y * 128;

  f32x8 acc[2][4];
#pragma unroll
  for (int i = 0; i < 2; ++i)
#pragma unroll
    for (int j = 0; j < 4; ++j) acc[i][j] = zero8();

  const int a_row = tid >> 1;               // 0..127
  const int a_kh  = (tid & 1) * 16;         // 0 or 16
  const int a_gm  = bm0 + a_row;
  const unsigned a_lds = (unsigned)(size_t)(As + a_row * 40 + a_kh);

  for (int k0 = 0; k0 < K; k0 += 32) {
    // ---- stage A tile: async global->LDS (32B per thread, 2 x b128) ----
    // Rows with a_gm >= M are left stale; their accumulator rows are dead
    // (store side is predicated on m_g < M), so no zero-fill is required.
    if (a_gm < M) {
      unsigned long long ga =
          (unsigned long long)(A + (size_t)a_gm * lda + k0 + a_kh);
      ASYNC_B128(a_lds, ga, 0);
      ASYNC_B128(a_lds, ga, 16);
      if (k0 + 32 < K)
        __builtin_prefetch(A + (size_t)a_gm * lda + k0 + 32 + a_kh, 0, 1);
    }
    // ---- stage B tile, transposed into Bt[n][k] (manual: needs transpose) ----
#pragma unroll
    for (int s = tid; s < 512; s += 256) {
      int kk = s >> 4;                       // 0..31
      int np = (s & 15) * 8;                 // 0..120
      const __bf16* gsrc = B + (size_t)(k0 + kk) * ldb + bn0 + np;
      uint4 u = *(const uint4*)gsrc;
      if (k0 + 32 < K) __builtin_prefetch(gsrc + (size_t)32 * ldb, 0, 1);
      const __bf16* e = (const __bf16*)&u;
#pragma unroll
      for (int j = 0; j < 8; ++j) Bt[(np + j) * 40 + kk] = e[j];
    }
    wait_asynccnt0();      // async A-tile writes visible in LDS
    __syncthreads();

    // ---- fragments + WMMA ----
    bf16x16 af[2], bfg[4];
    const int ks8 = (lane >> 4) * 8;
    const int kb  = (lane >> 4) * 16;
    const int m0  = wm * 32 + (lane & 15);
    const int n0  = wn * 64 + (lane & 15);
#pragma unroll
    for (int mt = 0; mt < 2; ++mt) {
      const __bf16* p = As + (m0 + mt * 16) * 40;
      af[mt] = make_frag(*(const uint4*)(p + ks8), *(const uint4*)(p + 16 + ks8));
    }
#pragma unroll
    for (int nt = 0; nt < 4; ++nt) {
      const __bf16* p = Bt + (n0 + nt * 16) * 40 + ks8;
      bfg[nt] = make_frag(*(const uint4*)p, *(const uint4*)(p + 16));
    }
#pragma unroll
    for (int mt = 0; mt < 2; ++mt)
#pragma unroll
      for (int nt = 0; nt < 4; ++nt)
        acc[mt][nt] = __builtin_amdgcn_wmma_f32_16x16x32_bf16(
            false, af[mt], false, bfg[nt], (short)0, acc[mt][nt], false, false);
    __syncthreads();
  }

  // ---- epilogue: bias + activation + alpha, fp32 or bf16 store ----
  const int n_base = bn0 + wn * 64;
#pragma unroll
  for (int nt = 0; nt < 4; ++nt) {
    int   n_g = n_base + nt * 16 + (lane & 15);
    float bv  = bias[n_g];
#pragma unroll
    for (int mt = 0; mt < 2; ++mt) {
      int m_loc = bm0 + wm * 32 + mt * 16 + (lane >> 4) * 8;
#pragma unroll
      for (int r = 0; r < 8; ++r) {
        int m_g = m_loc + r;
        if (m_g < M) {
          float v = (acc[mt][nt][r] + bv) * alpha;
          if (ACT == 1) v = v > 0.0f ? v : 0.0f;
          if (OUT_BF16)
            ((__bf16*)Cout)[(size_t)m_g * ldc + n_g] = (__bf16)v;
          else
            ((float*)Cout)[(size_t)m_g * ldc + n_g] = v;
        }
      }
    }
  }
}

// ---------------------------------------------------------------------------
// Fused cross-attention per (b, h): logits = q@k^T (+bias), softmax, ctx = P@v.
// 128 threads (4 waves); each wave handles 64 query rows (4 tiles of 16).
// q/k/v are bf16 with rows indexed (t*B + b), cols h*64 + d.
// ---------------------------------------------------------------------------
__global__ __launch_bounds__(128)
void attn_bf16_wmma_kernel(const __bf16* __restrict__ q,
                           const __bf16* __restrict__ k,
                           const __bf16* __restrict__ v,
                           const float* __restrict__ src_bias,
                           __bf16* __restrict__ ctx) {
  __shared__ __bf16 ks[SS * 72];        // Bt for logits: ks[s][d]
  __shared__ __bf16 vt[DD * 136];       // Bt for ctx   : vt[d][s]
  __shared__ __bf16 Pw[4][16 * 136];    // per-wave prob tile in A layout [m][s]

  const int b    = blockIdx.x >> 4;     // 0..31
  const int h    = blockIdx.x & 15;     // 0..15
  const int tid  = threadIdx.x;
  const int lane = tid & 31;
  const int wave = tid >> 5;

  // stage K (already in Bt layout): straight copy -> async global->LDS.
  // Row capacity 144B, payload 128B at deltas 0..112 on both sides.
  {
    const unsigned lds_dst = (unsigned)(size_t)(ks + tid * 72);
    const unsigned long long gsrc =
        (unsigned long long)(k + ((size_t)(tid * BB + b)) * HH + h * DD);
    ASYNC_B128(lds_dst, gsrc, 0);
    ASYNC_B128(lds_dst, gsrc, 16);
    ASYNC_B128(lds_dst, gsrc, 32);
    ASYNC_B128(lds_dst, gsrc, 48);
    ASYNC_B128(lds_dst, gsrc, 64);
    ASYNC_B128(lds_dst, gsrc, 80);
    ASYNC_B128(lds_dst, gsrc, 96);
    ASYNC_B128(lds_dst, gsrc, 112);
  }
  // stage V transposed (manual: transpose changes the layout)
  {
    const __bf16* src = v + ((size_t)(tid * BB + b)) * HH + h * DD;
#pragma unroll
    for (int j = 0; j < 8; ++j) {
      uint4 u = *(const uint4*)(src + j * 8);
      const __bf16* e = (const __bf16*)&u;
#pragma unroll
      for (int jj = 0; jj < 8; ++jj) vt[(j * 8 + jj) * 136 + tid] = e[jj];
    }
  }
  wait_asynccnt0();
  __syncthreads();

  const int ks8 = (lane >> 4) * 8;
  const int kb  = (lane >> 4) * 16;

  for (int qt = 0; qt < 4; ++qt) {
    const int t0   = wave * 64 + qt * 16;
    const int trow = t0 + (lane & 15);

    // Q fragments straight from global (K=64 -> two 16x32 fragments)
    const __bf16* qrow = q + ((size_t)(trow * BB + b)) * HH + h * DD;
    bf16x16 aq[2];
#pragma unroll
    for (int kc = 0; kc < 2; ++kc)
      aq[kc] = make_frag(*(const uint4*)(qrow + kc * 32 + ks8),
                         *(const uint4*)(qrow + kc * 32 + 16 + ks8));

    // logits: 16 x 128
    f32x8 accl[8];
#pragma unroll
    for (int i = 0; i < 8; ++i) accl[i] = zero8();
#pragma unroll
    for (int nt = 0; nt < 8; ++nt)
#pragma unroll
      for (int kc = 0; kc < 2; ++kc) {
        const __bf16* p = ks + (nt * 16 + (lane & 15)) * 72 + kc * 32 + ks8;
        bf16x16 bk = make_frag(*(const uint4*)p, *(const uint4*)(p + 16));
        accl[nt] = __builtin_amdgcn_wmma_f32_16x16x32_bf16(
            false, aq[kc], false, bk, (short)0, accl[nt], false, false);
      }

    // bias + softmax along S (rows split across lane halves; xor<16 reductions)
    float bias_v[8];
#pragma unroll
    for (int nt = 0; nt < 8; ++nt)
      bias_v[nt] = src_bias[b * SS + nt * 16 + (lane & 15)];

    float rinv[8];
#pragma unroll
    for (int r = 0; r < 8; ++r) {
      float m = -3.0e38f;
#pragma unroll
      for (int nt = 0; nt < 8; ++nt) {
        float x = accl[nt][r] + bias_v[nt];
        accl[nt][r] = x;
        m = fmaxf(m, x);
      }
#pragma unroll
      for (int off = 1; off < 16; off <<= 1) m = fmaxf(m, __shfl_xor(m, off, 32));
      float s = 0.0f;
#pragma unroll
      for (int nt = 0; nt < 8; ++nt) {
        float e = __expf(accl[nt][r] - m);
        accl[nt][r] = e;
        s += e;
      }
#pragma unroll
      for (int off = 1; off < 16; off <<= 1) s += __shfl_xor(s, off, 32);
      rinv[r] = 1.0f / s;
    }

    // write probabilities into per-wave LDS tile in A layout [m][s]
    __bf16* P = &Pw[wave][0];
#pragma unroll
    for (int r = 0; r < 8; ++r) {
      int row = r + (lane >> 4) * 8;
#pragma unroll
      for (int nt = 0; nt < 8; ++nt)
        P[row * 136 + nt * 16 + (lane & 15)] = (__bf16)(accl[nt][r] * rinv[r]);
    }

    // ctx = P(16x128) @ V(128x64)
    f32x8 accc[4];
#pragma unroll
    for (int i = 0; i < 4; ++i) accc[i] = zero8();
#pragma unroll
    for (int kc = 0; kc < 4; ++kc) {
      const __bf16* pa = P + (lane & 15) * 136 + kc * 32 + ks8;
      bf16x16 ap = make_frag(*(const uint4*)pa, *(const uint4*)(pa + 16));
#pragma unroll
      for (int nt = 0; nt < 4; ++nt) {
        const __bf16* pb = vt + (nt * 16 + (lane & 15)) * 136 + kc * 32 + ks8;
        bf16x16 bv = make_frag(*(const uint4*)pb, *(const uint4*)(pb + 16));
        accc[nt] = __builtin_amdgcn_wmma_f32_16x16x32_bf16(
            false, ap, false, bv, (short)0, accc[nt], false, false);
      }
    }

    // scatter ctx (rows (t*B+b), cols h*64+d)
#pragma unroll
    for (int nt = 0; nt < 4; ++nt) {
      int d = nt * 16 + (lane & 15);
#pragma unroll
      for (int r = 0; r < 8; ++r) {
        int tg = t0 + r + (lane >> 4) * 8;
        ctx[((size_t)(tg * BB + b)) * HH + h * DD + d] = (__bf16)accc[nt][r];
      }
    }
  }
}

// ---------------------------------------------------------------------------
// LSTM cell: G[32,4096] (i|j|f|o) -> new c/h; h also fed back (bf16) into xs[t+1]
// ---------------------------------------------------------------------------
__global__ __launch_bounds__(256)
void lstm_cell_kernel(const float* __restrict__ G, float* __restrict__ c,
                      __bf16* __restrict__ xs, float* __restrict__ out, int t) {
  int idx = blockIdx.x * 256 + threadIdx.x;     // 0 .. 32767
  int rb  = idx >> 10;
  int col = idx & 1023;
  const float* g = G + (size_t)rb * 4096;
  float gi = g[col];
  float gj = g[1024 + col];
  float gf = g[2048 + col];
  float go = g[3072 + col];
  float i_ = 1.0f / (1.0f + __expf(-gi));
  float j_ = tanhf(gj);
  float f_ = 1.0f / (1.0f + __expf(-gf));
  float o_ = 1.0f / (1.0f + __expf(-go));
  float nc = f_ * c[idx] + i_ * j_;
  float nh = o_ * nc;                           // THUMT LSTMCell: activation=None
  c[idx] = nc;
  out[(size_t)t * (BB * HH) + idx] = nh;        // hiddens[t]
  if (t + 1 < TT) {
    xs[((size_t)(t + 1) * BB + rb) * 3072 + 2048 + col] = (__bf16)nh;
  } else {
    out[(size_t)TT * BB * HH + idx] = nc;              // c_f
    out[(size_t)TT * BB * HH + BB * HH + idx] = nh;    // h_f
  }
}

// ---------------------------------------------------------------------------
// Host orchestration
// ---------------------------------------------------------------------------
extern "C" void kernel_launch(void* const* d_in, const int* in_sizes, int n_in,
                              void* d_out, int out_size, void* d_ws, size_t ws_size,
                              hipStream_t stream) {
  const float* x        = (const float*)d_in[0];
  const float* src_bias = (const float*)d_in[1];
  // d_in[2] = tgt_bias (unused by the reference computation)
  const float* memory   = (const float*)d_in[3];
  const float* Wq = (const float*)d_in[4];
  const float* bq = (const float*)d_in[5];
  const float* Wk = (const float*)d_in[6];
  const float* bk = (const float*)d_in[7];
  const float* Wv = (const float*)d_in[8];
  const float* bv = (const float*)d_in[9];
  const float* Wo = (const float*)d_in[10];
  const float* bo = (const float*)d_in[11];
  const float* W1 = (const float*)d_in[12];
  const float* b1 = (const float*)d_in[13];
  const float* W2 = (const float*)d_in[14];
  const float* b2 = (const float*)d_in[15];
  const float* Wg = (const float*)d_in[16];
  const float* bg = (const float*)d_in[17];
  float* out = (float*)d_out;

  // --- workspace carve-up (bytes, 256-aligned) ---
  char*  ws  = (char*)d_ws;
  size_t off = 0;
  auto alloc = [&](size_t bytes) -> void* {
    void* p = ws + off;
    off = (off + bytes + 255) & ~(size_t)255;
    return p;
  };
  const size_t TBH = (size_t)TT * BB * HH;      // 8,388,608
  const size_t SBH = (size_t)SS * BB * HH;      // 4,194,304

  __bf16* x_bf   = (__bf16*)alloc(TBH * 2);
  __bf16* mem_bf = (__bf16*)alloc(SBH * 2);
  __bf16* wq_bf  = (__bf16*)alloc((size_t)HH * HH * 2);
  __bf16* wk_bf  = (__bf16*)alloc((size_t)HH * HH * 2);
  __bf16* wv_bf  = (__bf16*)alloc((size_t)HH * HH * 2);
  __bf16* wo_bf  = (__bf16*)alloc((size_t)HH * HH * 2);
  __bf16* w1_bf  = (__bf16*)alloc((size_t)HH * FF * 2);
  __bf16* w2_bf  = (__bf16*)alloc((size_t)FF * HH * 2);
  __bf16* wg_bf  = (__bf16*)alloc((size_t)3 * HH * 4 * HH * 2);
  __bf16* hid_bf = (__bf16*)alloc(TBH * FF / HH * 2);   // [8192,4096] bf16 = 64MB
  __bf16* xs     = (__bf16*)alloc((size_t)TT * BB * 3 * HH * 2);
  float*  G      = (float*)alloc((size_t)BB * 4 * HH * 4);
  float*  c      = (float*)alloc((size_t)BB * HH * 4);
  // q/k/v/ctx alias hid_bf (FFN runs first; 48MB fits in the 64MB region)
  __bf16* q_bf   = hid_bf;
  __bf16* k_bf   = hid_bf + TBH;
  __bf16* v_bf   = k_bf + SBH;
  __bf16* ctx_bf = v_bf + SBH;

  auto cvt = [&](const float* src, __bf16* dst, size_t n) {
    cvt_f32_bf16_kernel<<<(unsigned)((n + 255) / 256), 256, 0, stream>>>(src, dst, (int)n);
  };
  auto gemm = [&](const __bf16* A, int lda, const __bf16* B, int ldb,
                  const float* bias, float alpha, void* C, int ldc,
                  int M, int N, int K, int act, bool outbf) {
    dim3 grid(N / 128, (M + 127) / 128);
    if (outbf) {
      if (act)
        gemm_bf16_wmma_kernel<1, true><<<grid, 256, 0, stream>>>(A, lda, B, ldb, bias, alpha, C, ldc, M, N, K);
      else
        gemm_bf16_wmma_kernel<0, true><<<grid, 256, 0, stream>>>(A, lda, B, ldb, bias, alpha, C, ldc, M, N, K);
    } else {
      gemm_bf16_wmma_kernel<0, false><<<grid, 256, 0, stream>>>(A, lda, B, ldb, bias, alpha, C, ldc, M, N, K);
    }
  };

  // --- 1. precision conversion ---
  cvt(x, x_bf, TBH);
  cvt(memory, mem_bf, SBH);
  cvt(Wq, wq_bf, (size_t)HH * HH);
  cvt(Wk, wk_bf, (size_t)HH * HH);
  cvt(Wv, wv_bf, (size_t)HH * HH);
  cvt(Wo, wo_bf, (size_t)HH * HH);
  cvt(W1, w1_bf, (size_t)HH * FF);
  cvt(W2, w2_bf, (size_t)FF * HH);
  cvt(Wg, wg_bf, (size_t)3 * HH * 4 * HH);

  // --- 2. init LSTM state (c0 = 0, h0 = 0 in xs[0]) ---
  init_state_kernel<<<(BB * HH) / 256, 256, 0, stream>>>(c, xs);

  // --- 3. FFN first (so its hidden buffer can be reused for q/k/v/ctx) ---
  gemm(x_bf, HH, w1_bf, FF, b1, 1.0f, hid_bf, FF, TT * BB, FF, HH, /*relu*/1, true);
  gemm(hid_bf, FF, w2_bf, HH, b2, 1.0f, xs + 0, 3 * HH, TT * BB, HH, FF, 0, true);

  // --- 4. attention ---
  const float qscale = 0.125f;  // D^-0.5, D=64
  gemm(x_bf, HH, wq_bf, HH, bq, qscale, q_bf, HH, TT * BB, HH, HH, 0, true);
  gemm(mem_bf, HH, wk_bf, HH, bk, 1.0f, k_bf, HH, SS * BB, HH, HH, 0, true);
  gemm(mem_bf, HH, wv_bf, HH, bv, 1.0f, v_bf, HH, SS * BB, HH, HH, 0, true);
  attn_bf16_wmma_kernel<<<BB * NHH, 128, 0, stream>>>(q_bf, k_bf, v_bf, src_bias, ctx_bf);
  gemm(ctx_bf, HH, wo_bf, HH, bo, 1.0f, xs + HH, 3 * HH, TT * BB, HH, HH, 0, true);

  // --- 5. LSTM scan: 256 stream-ordered (gates GEMM, cell) pairs ---
  for (int t = 0; t < TT; ++t) {
    gemm(xs + (size_t)t * BB * 3 * HH, 3 * HH, wg_bf, 4 * HH, bg, 1.0f,
         G, 4 * HH, BB, 4 * HH, 3 * HH, 0, false);
    lstm_cell_kernel<<<(BB * HH) / 256, 256, 0, stream>>>(G, c, xs, out, t);
  }

  (void)in_sizes; (void)n_in; (void)out_size; (void)ws_size;
}
